// SwinBlock_53283364274261
// MI455X (gfx1250) — matrix operand order repaired and checked by reading, not compiled, not run on hardware
//
#include <hip/hip_runtime.h>
#include <hip/hip_bf16.h>
#include <math.h>

// ---------------- constants ----------------
constexpr int BATCH = 64;
constexpr int HH    = 28;
constexpr int WW    = 28;
constexpr int DIM   = 384;
constexpr int HEADS = 12;
constexpr int HD    = 32;
constexpr int WIN   = 7;
constexpr int SHIFT = 3;
constexpr int NTOK  = 49;                 // tokens per window
constexpr int NWIMG = 16;                 // windows per image
constexpr int NWIN  = BATCH * NWIMG;      // 1024 windows total
constexpr int MTOT  = NWIN * NTOK;        // 50176 rows
constexpr int QKVN  = 3 * DIM;            // 1152
constexpr int HID   = 4 * DIM;            // 1536
constexpr int NIMG  = HH * WW;            // 784
constexpr float SCALE = 0.17677669529663687f; // 32^-0.5

// ---------------- WMMA types ----------------
typedef __attribute__((ext_vector_type(16))) __bf16 v16bf;
typedef __attribute__((ext_vector_type(8)))  float  v8f;

#define WMMA_BF16(a, b, c) \
  __builtin_amdgcn_wmma_f32_16x16x32_bf16(false, (a), false, (b), (short)0, (c), false, false)

static __device__ __forceinline__ v16bf ldfrag(const __bf16* p0, const __bf16* p1) {
  union { uint4 u[2]; v16bf v; } f;
  f.u[0] = *reinterpret_cast<const uint4*>(p0);
  f.u[1] = *reinterpret_cast<const uint4*>(p1);
  return f.v;
}

// windowed-token -> original-token index (handles shift + window partition/reverse)
static __device__ __forceinline__ int remap_row(int t) {
  int b   = t / NIMG;
  int rem = t - b * NIMG;
  int wi  = rem / NTOK;
  int m   = rem - wi * NTOK;
  int wr = wi >> 2, wc = wi & 3;
  int r = m / WIN, c = m - r * WIN;
  int hs = wr * WIN + r, ws = wc * WIN + c;
  int h0 = hs + SHIFT; if (h0 >= HH) h0 -= HH;
  int w0 = ws + SHIFT; if (w0 >= WW) w0 -= WW;
  return b * NIMG + h0 * WW + w0;
}

static __device__ __forceinline__ int region9(int z) {
  return (z < HH - WIN) ? 0 : ((z < HH - SHIFT) ? 1 : 2);
}

// ---------------- weight f32 -> bf16 conversion ----------------
__global__ void cvt_weights(const float* __restrict__ qkv_w, const float* __restrict__ proj_w,
                            const float* __restrict__ fc1_w, const float* __restrict__ fc2_w,
                            __bf16* __restrict__ wq, __bf16* __restrict__ wp,
                            __bf16* __restrict__ w1, __bf16* __restrict__ w2) {
  const int nq = QKVN * DIM, np = DIM * DIM, n1 = HID * DIM, n2 = DIM * HID;
  const int ntot = nq + np + n1 + n2;
  for (int i = blockIdx.x * blockDim.x + threadIdx.x; i < ntot; i += gridDim.x * blockDim.x) {
    if (i < nq)                wq[i]            = (__bf16)qkv_w[i];
    else if (i < nq + np)      wp[i - nq]       = (__bf16)proj_w[i - nq];
    else if (i < nq + np + n1) w1[i - nq - np]  = (__bf16)fc1_w[i - nq - np];
    else                       w2[i - nq - np - n1] = (__bf16)fc2_w[i - nq - np - n1];
  }
}

// ---------------- layernorm (optionally with shift+partition remap) ----------------
template <bool REMAP>
__global__ __launch_bounds__(256) void ln_kernel(const float* __restrict__ x,
                                                 const float* __restrict__ gw,
                                                 const float* __restrict__ gb,
                                                 __bf16* __restrict__ out) {
  const int lane = threadIdx.x & 31;
  const int wv   = threadIdx.x >> 5;
  const int t    = blockIdx.x * 8 + wv;
  const int src  = REMAP ? remap_row(t) : t;
  const float* xr = x + (size_t)src * DIM;

  float vals[12];
  float s = 0.f;
#pragma unroll
  for (int i = 0; i < 12; ++i) { vals[i] = xr[lane + 32 * i]; s += vals[i]; }
#pragma unroll
  for (int m = 1; m < 32; m <<= 1) s += __shfl_xor(s, m, 32);
  const float mu = s * (1.0f / DIM);
  float vs = 0.f;
#pragma unroll
  for (int i = 0; i < 12; ++i) { float d = vals[i] - mu; vs += d * d; }
#pragma unroll
  for (int m = 1; m < 32; m <<= 1) vs += __shfl_xor(vs, m, 32);
  const float rstd = rsqrtf(vs * (1.0f / DIM) + 1e-5f);

  __bf16* orow = out + (size_t)t * DIM;
#pragma unroll
  for (int i = 0; i < 12; ++i) {
    int idx = lane + 32 * i;
    orow[idx] = (__bf16)((vals[i] - mu) * rstd * gw[idx] + gb[idx]);
  }
}

// ---------------- generic WMMA GEMM: C = A(MxK) * W(NxK)^T + bias ----------------
// Each wave computes a 16x128 output slab (8 WMMA tiles per K step from one A frag).
// MODE 0: store bf16 ; 1: gelu->bf16 ; 2: f32 + res (identity rows, out=f32) ;
// MODE 3: f32 + res with window-reverse row remap (proj epilogue)
template <int MODE>
__global__ __launch_bounds__(128) void gemm_bf16(const __bf16* __restrict__ A,
                                                 const __bf16* __restrict__ W,
                                                 const float* __restrict__ bias,
                                                 int M, int N, int K,
                                                 void* __restrict__ outp,
                                                 const float* __restrict__ res) {
  const int lane = threadIdx.x & 31;
  const int wv   = threadIdx.x >> 5;
  const int g    = lane >> 4;
  const int ln   = lane & 15;
  const int rowBase = blockIdx.y * 64 + wv * 16;
  const int colBase = blockIdx.x * 128;

  v8f acc[8];
#pragma unroll
  for (int nt = 0; nt < 8; ++nt) acc[nt] = {};

  const __bf16* arow  = A + (size_t)(rowBase + ln) * K;
  const __bf16* wbase = W + (size_t)(colBase + ln) * K + 16 * g;
  const size_t  wstep = (size_t)16 * K;   // 16 weight rows per n-tile

  for (int k0 = 0; k0 < K; k0 += 32) {
    // A frag: contiguous {k0+16g .. +15}
    v16bf a = ldfrag(arow + k0 + 16 * g, arow + k0 + 16 * g + 8);
#pragma unroll
    for (int nt = 0; nt < 8; ++nt) {
      const __bf16* wp = wbase + (size_t)nt * wstep + k0;
      v16bf b = ldfrag(wp, wp + 8);
      acc[nt] = WMMA_BF16(a, b, acc[nt]);
    }
  }

#pragma unroll
  for (int r = 0; r < 8; ++r) {
    const int row = rowBase + 8 * g + r;
    int orow = row;
    if constexpr (MODE == 3) orow = remap_row(row);
#pragma unroll
    for (int nt = 0; nt < 8; ++nt) {
      const int col = colBase + nt * 16 + ln;
      float v = acc[nt][r] + bias[col];
      if constexpr (MODE == 0) {
        ((__bf16*)outp)[(size_t)row * N + col] = (__bf16)v;
      } else if constexpr (MODE == 1) {
        v = 0.5f * v * (1.0f + erff(v * 0.70710678118654752f));
        ((__bf16*)outp)[(size_t)row * N + col] = (__bf16)v;
      } else if constexpr (MODE == 2) {
        size_t oi = (size_t)row * N + col;
        ((float*)outp)[oi] = v + res[oi];
      } else {
        size_t oi = (size_t)orow * N + col;
        ((float*)outp)[oi] = v + res[oi];
      }
    }
  }
}

// ---------------- V repack: [win,h,d,64(pad m)] bf16, zero padded ----------------
__global__ __launch_bounds__(256) void repack_v(const __bf16* __restrict__ qkv,
                                                __bf16* __restrict__ vt) {
  int idx = blockIdx.x * blockDim.x + threadIdx.x; // NWIN*HEADS*HD*64 total
  int m  = idx & 63;
  int d  = (idx >> 6) & 31;
  int wh = idx >> 11;
  int h  = wh % HEADS;
  int w  = wh / HEADS;
  __bf16 v = (__bf16)0.0f;
  if (m < NTOK) v = qkv[(size_t)(w * NTOK + m) * QKVN + 2 * DIM + h * HD + d];
  vt[idx] = v;
}

// ---------------- attention: one wave = one (window, head) ----------------
__global__ __launch_bounds__(128) void attn_kernel(const __bf16* __restrict__ qkv,
                                                   const __bf16* __restrict__ vt,
                                                   const float* __restrict__ bias_table,
                                                   __bf16* __restrict__ obuf) {
  __shared__ __align__(16) __bf16 P[4][64][64];   // per-wave softmax probs (bf16)
  __shared__ float btabh[4][172];                 // per-wave head-specific bias row

  const int lane = threadIdx.x & 31;
  const int wv   = threadIdx.x >> 5;
  const int g    = lane >> 4;
  const int ln   = lane & 15;
  const int pair = blockIdx.x * 4 + wv;
  const int w    = pair / HEADS;
  const int h    = pair % HEADS;
  const int wi   = w & (NWIMG - 1);
  const int wr7  = (wi >> 2) * WIN;
  const int wc7  = (wi & 3) * WIN;

  // stage this head's 169 bias values (indexed by mcode-ncode+84)
  for (int i = lane; i < 169; i += 32) btabh[wv][i] = bias_table[i * HEADS + h];
  __syncthreads();

  // per-lane column-side precompute (independent of the row loop)
  int ncode[4], nrid[4];
  bool nval[4];
#pragma unroll
  for (int nt = 0; nt < 4; ++nt) {
    const int n = nt * 16 + ln;
    nval[nt] = (n < NTOK);
    const int nn = nval[nt] ? n : NTOK - 1;
    const int nr = nn / WIN, nc = nn - nr * WIN;
    ncode[nt] = nr * 13 + nc;
    nrid[nt]  = region9(wr7 + nr) * 3 + region9(wc7 + nc);
  }

  // K-operand fragments (B matrix of Q*K^T): row n of K-matrix is contiguous (head dim)
  v16bf kf[4];
#pragma unroll
  for (int nt = 0; nt < 4; ++nt) {
    const int n = nt * 16 + ln;
    const int tok = (n < NTOK) ? n : NTOK - 1;
    const __bf16* kp = qkv + (size_t)(w * NTOK + tok) * QKVN + DIM + h * HD + 16 * g;
    kf[nt] = ldfrag(kp, kp + 8);
  }

#pragma unroll 1
  for (int mt = 0; mt < 4; ++mt) {
    const int ma  = mt * 16 + ln;
    const int tka = (ma < NTOK) ? ma : NTOK - 1;
    const __bf16* qp = qkv + (size_t)(w * NTOK + tka) * QKVN + h * HD;
    v16bf af = ldfrag(qp + 16 * g, qp + 16 * g + 8);

    v8f sc[4];
    sc[0] = {}; sc[1] = {}; sc[2] = {}; sc[3] = {};
    sc[0] = WMMA_BF16(af, kf[0], sc[0]);
    sc[1] = WMMA_BF16(af, kf[1], sc[1]);
    sc[2] = WMMA_BF16(af, kf[2], sc[2]);
    sc[3] = WMMA_BF16(af, kf[3], sc[3]);

    // branch-free softmax in accumulator layout (row m = mt*16+8g+r, col n = nt*16+ln)
#pragma unroll
    for (int r = 0; r < 8; ++r) {
      const int m    = mt * 16 + 8 * g + r;
      const bool mval = (m < NTOK);
      const int mm   = mval ? m : NTOK - 1;
      const int mr   = mm / WIN, mc = mm - mr * WIN;
      const int mcode = mr * 13 + mc + 84;
      const int mrid  = region9(wr7 + mr) * 3 + region9(wc7 + mc);

      float sv[4];
#pragma unroll
      for (int nt = 0; nt < 4; ++nt) {
        float b   = btabh[wv][mcode - ncode[nt]];
        float val = fmaf(sc[nt][r], SCALE, b) + ((mrid != nrid[nt]) ? -100.0f : 0.0f);
        val = mval ? val : 0.0f;          // dead rows: keep finite
        sv[nt] = nval[nt] ? val : -1e30f; // dead cols: exclude from softmax
      }
      float rmax = fmaxf(fmaxf(sv[0], sv[1]), fmaxf(sv[2], sv[3]));
#pragma unroll
      for (int msk = 1; msk < 16; msk <<= 1) rmax = fmaxf(rmax, __shfl_xor(rmax, msk, 32));
      float e[4];
      float rsum = 0.f;
#pragma unroll
      for (int nt = 0; nt < 4; ++nt) {
        e[nt] = nval[nt] ? __expf(sv[nt] - rmax) : 0.0f;
        rsum += e[nt];
      }
#pragma unroll
      for (int msk = 1; msk < 16; msk <<= 1) rsum += __shfl_xor(rsum, msk, 32);
      const float inv = 1.0f / rsum;
#pragma unroll
      for (int nt = 0; nt < 4; ++nt) P[wv][m][nt * 16 + ln] = (__bf16)(e[nt] * inv);
    }
  }
  __syncthreads();

  // O = P(64x64) @ V(64x32): B operand from pre-transposed, zero-padded V^T
  v16bf vf[2][2];
#pragma unroll
  for (int dt = 0; dt < 2; ++dt)
#pragma unroll
    for (int kk = 0; kk < 2; ++kk) {
      const __bf16* vp = vt + ((size_t)(w * HEADS + h) * HD + dt * 16 + ln) * 64 + kk * 32 + 16 * g;
      vf[dt][kk] = ldfrag(vp, vp + 8);
    }

#pragma unroll 1
  for (int mt = 0; mt < 4; ++mt) {
    v8f o0 = {}, o1 = {};
#pragma unroll
    for (int kk = 0; kk < 2; ++kk) {
      const __bf16* pp = &P[wv][mt * 16 + ln][0] + kk * 32 + 16 * g;
      v16bf pa = ldfrag(pp, pp + 8);
      o0 = WMMA_BF16(pa, vf[0][kk], o0);
      o1 = WMMA_BF16(pa, vf[1][kk], o1);
    }
#pragma unroll
    for (int r = 0; r < 8; ++r) {
      const int m = mt * 16 + 8 * g + r;
      if (m < NTOK) {
        size_t rb = (size_t)(w * NTOK + m) * DIM + h * HD;
        obuf[rb + ln]      = (__bf16)o0[r];
        obuf[rb + 16 + ln] = (__bf16)o1[r];
      }
    }
  }
}

// ---------------- workspace layout (bytes) ----------------
constexpr size_t OFF_XW   = 0;                                        // XW bf16 / Z bf16
constexpr size_t SZ_XW    = (size_t)MTOT * DIM * 2;                   // 38.5 MB
constexpr size_t OFF_QKV  = OFF_XW + SZ_XW;                           // QKV bf16 / MLP hidden bf16
constexpr size_t SZ_QKV   = (size_t)MTOT * HID * 2;                   // 154.1 MB (covers both)
constexpr size_t OFF_VT   = OFF_QKV + SZ_QKV;                         // V^T bf16
constexpr size_t SZ_VT    = (size_t)NWIN * HEADS * HD * 64 * 2;       // 50.3 MB
constexpr size_t OFF_OB   = OFF_VT + SZ_VT;                           // attention out bf16
constexpr size_t SZ_OB    = (size_t)MTOT * DIM * 2;
constexpr size_t OFF_XRES = OFF_OB + SZ_OB;                           // residual f32
constexpr size_t SZ_XRES  = (size_t)MTOT * DIM * 4;
constexpr size_t OFF_W    = OFF_XRES + SZ_XRES;                       // bf16 weights
constexpr size_t OFF_WQ   = OFF_W;
constexpr size_t OFF_WP   = OFF_WQ + (size_t)QKVN * DIM * 2;
constexpr size_t OFF_W1   = OFF_WP + (size_t)DIM * DIM * 2;
constexpr size_t OFF_W2   = OFF_W1 + (size_t)HID * DIM * 2;

extern "C" void kernel_launch(void* const* d_in, const int* in_sizes, int n_in,
                              void* d_out, int out_size, void* d_ws, size_t ws_size,
                              hipStream_t stream) {
  (void)in_sizes; (void)n_in; (void)out_size; (void)ws_size;
  const float* x      = (const float*)d_in[0];
  const float* n1w    = (const float*)d_in[1];
  const float* n1b    = (const float*)d_in[2];
  const float* qkv_w  = (const float*)d_in[3];
  const float* qkv_b  = (const float*)d_in[4];
  const float* proj_w = (const float*)d_in[5];
  const float* proj_b = (const float*)d_in[6];
  const float* btbl   = (const float*)d_in[7];
  const float* n2w    = (const float*)d_in[8];
  const float* n2b    = (const float*)d_in[9];
  const float* fc1_w  = (const float*)d_in[10];
  const float* fc1_b  = (const float*)d_in[11];
  const float* fc2_w  = (const float*)d_in[12];
  const float* fc2_b  = (const float*)d_in[13];

  char* ws = (char*)d_ws;
  __bf16* XW   = (__bf16*)(ws + OFF_XW);
  __bf16* QKV  = (__bf16*)(ws + OFF_QKV);
  __bf16* HB   = (__bf16*)(ws + OFF_QKV);   // aliases QKV (dead by then)
  __bf16* VT   = (__bf16*)(ws + OFF_VT);
  __bf16* OB   = (__bf16*)(ws + OFF_OB);
  float*  XRES = (float*)(ws + OFF_XRES);
  __bf16* Z    = (__bf16*)(ws + OFF_XW);    // aliases XW (dead by then)
  __bf16* WQ   = (__bf16*)(ws + OFF_WQ);
  __bf16* WP   = (__bf16*)(ws + OFF_WP);
  __bf16* W1   = (__bf16*)(ws + OFF_W1);
  __bf16* W2   = (__bf16*)(ws + OFF_W2);
  float*  OUT  = (float*)d_out;

  // 1. weights -> bf16
  cvt_weights<<<2048, 256, 0, stream>>>(qkv_w, proj_w, fc1_w, fc2_w, WQ, WP, W1, W2);
  // 2. LN1 + shift + window partition -> XW
  ln_kernel<true><<<MTOT / 8, 256, 0, stream>>>(x, n1w, n1b, XW);
  // 3. QKV GEMM
  gemm_bf16<0><<<dim3(QKVN / 128, MTOT / 64), 128, 0, stream>>>(
      XW, WQ, qkv_b, MTOT, QKVN, DIM, (void*)QKV, nullptr);
  // 4. V transpose/pad
  repack_v<<<(NWIN * HEADS * HD * 64) / 256, 256, 0, stream>>>(QKV, VT);
  // 5. windowed attention
  attn_kernel<<<(NWIN * HEADS) / 4, 128, 0, stream>>>(QKV, VT, btbl, OB);
  // 6. proj GEMM + window reverse + unshift + residual -> XRES
  gemm_bf16<3><<<dim3(DIM / 128, MTOT / 64), 128, 0, stream>>>(
      OB, WP, proj_b, MTOT, DIM, DIM, (void*)XRES, x);
  // 7. LN2 -> Z
  ln_kernel<false><<<MTOT / 8, 256, 0, stream>>>(XRES, n2w, n2b, Z);
  // 8. fc1 GEMM + exact GELU
  gemm_bf16<1><<<dim3(HID / 128, MTOT / 64), 128, 0, stream>>>(
      Z, W1, fc1_b, MTOT, HID, DIM, (void*)HB, nullptr);
  // 9. fc2 GEMM + residual -> d_out
  gemm_bf16<2><<<dim3(DIM / 128, MTOT / 64), 128, 0, stream>>>(
      HB, W2, fc2_b, MTOT, DIM, HID, (void*)OUT, XRES);
}
